// RWKV_Block_12893491822596
// MI455X (gfx1250) — compile-verified, matrix operand order checked by reading, not run on hardware
//
#include <hip/hip_runtime.h>
#include <hip/hip_bf16.h>

#define BB    2
#define TT    256
#define DD    2048
#define HH    32
#define HSZ   64
#define NTOK  (BB*TT)       // 512
#define LORAR 32
#define DLORA 64
#define STR   66
#define EPSV  1e-5f

typedef __bf16 bf16;
typedef __attribute__((ext_vector_type(16))) __bf16 v16bf;
typedef __attribute__((ext_vector_type(8)))  float  v8f;
typedef __attribute__((ext_vector_type(4)))  unsigned int v4u;
typedef __attribute__((ext_vector_type(8)))  int v8i_t;
typedef __attribute__((ext_vector_type(4)))  int v4i_t;

struct __align__(16) U4 { unsigned int x, y, z, w; };
union FragA { U4 u[2]; v16bf v; };

// ---------------------------------------------------------------------------
// fp32 -> bf16 conversion (weights / operands), n multiple of 4
// ---------------------------------------------------------------------------
__global__ __launch_bounds__(256)
void cvt_bf16_kernel(const float* __restrict__ src, bf16* __restrict__ dst, int n) {
    int i = (blockIdx.x * 256 + threadIdx.x) * 4;
    if (i < n) {
        float4 f = *(const float4*)(src + i);
        dst[i + 0] = (bf16)f.x;
        dst[i + 1] = (bf16)f.y;
        dst[i + 2] = (bf16)f.z;
        dst[i + 3] = (bf16)f.w;
    }
}

// ---------------------------------------------------------------------------
// Per-token prep: LayerNorm, token shift, maa LoRA -> x5 (bf16 GEMM operands),
// decay LoRA -> w = exp(-exp(...)) (fp32).  One block per token, 256 threads.
// ---------------------------------------------------------------------------
__global__ __launch_bounds__(256)
void prep_kernel(const float* __restrict__ x, const float* __restrict__ state,
                 const float* __restrict__ ln1_w, const float* __restrict__ ln1_b,
                 const float* __restrict__ tmx, const float* __restrict__ w1,
                 const float* __restrict__ w2,
                 const float* __restrict__ maak, const float* __restrict__ maaw,
                 const float* __restrict__ maav, const float* __restrict__ maar,
                 const float* __restrict__ maag,
                 const float* __restrict__ tdecay, const float* __restrict__ wd1,
                 const float* __restrict__ wd2, const int* __restrict__ ip,
                 bf16* __restrict__ X5,      // [5][NTOK][DD]  (f=1 slot unused)
                 float* __restrict__ wout)   // [NTOK][DD]
{
    __shared__ float xs[DD];      // ln(x_t)
    __shared__ float sxs[DD];     // sx
    __shared__ float xxs[DD];     // xxx (reused later for x5[1])
    __shared__ float t5s[5 * LORAR];
    __shared__ float decs[DLORA];
    __shared__ float red[512];

    const int n   = blockIdx.x;
    const int b   = n / TT;
    const int t   = n % TT;
    const int tid = threadIdx.x;
    const int i1  = STR * ip[0] + 1;

    const float* xc = x + (size_t)n * DD;
    const float* xp = xc - DD;   // valid only when t > 0

    float s0 = 0.f, s1 = 0.f, p0 = 0.f, p1 = 0.f;
    for (int j = 0; j < 8; ++j) {
        int c = tid + 256 * j;
        float v = xc[c]; s0 += v; s1 += v * v;
        if (t > 0) { float u = xp[c]; p0 += u; p1 += u * u; }
    }
    red[tid] = s0; red[256 + tid] = s1; __syncthreads();
    for (int off = 128; off > 0; off >>= 1) {
        if (tid < off) { red[tid] += red[tid + off]; red[256 + tid] += red[256 + tid + off]; }
        __syncthreads();
    }
    float mean_c = red[0] / DD;
    float var_c  = red[256] / DD - mean_c * mean_c;
    __syncthreads();

    float mean_p = 0.f, var_p = 1.f;
    if (t > 0) {
        red[tid] = p0; red[256 + tid] = p1; __syncthreads();
        for (int off = 128; off > 0; off >>= 1) {
            if (tid < off) { red[tid] += red[tid + off]; red[256 + tid] += red[256 + tid + off]; }
            __syncthreads();
        }
        mean_p = red[0] / DD;
        var_p  = red[256] / DD - mean_p * mean_p;
        __syncthreads();
    }
    const float rsc = rsqrtf(var_c + EPSV);
    const float rsp = rsqrtf(var_p + EPSV);

    for (int j = 0; j < 8; ++j) {
        int c = tid + 256 * j;
        float wv = ln1_w[c], bv = ln1_b[c];
        float xl = (xc[c] - mean_c) * rsc * wv + bv;
        float xprev = (t > 0) ? ((xp[c] - mean_p) * rsp * wv + bv)
                              : state[((size_t)b * STR + i1) * DD + c];
        float sx = xprev - xl;
        xs[c]  = xl;
        sxs[c] = sx;
        xxs[c] = xl + sx * tmx[c];
    }
    __syncthreads();

    // t5 = tanh(xxx @ W1),  W1: [DD, 160]
    if (tid < 5 * LORAR) {
        float acc = 0.f;
        for (int c = 0; c < DD; ++c) acc += xxs[c] * w1[(size_t)c * (5 * LORAR) + tid];
        t5s[tid] = tanhf(acc);
    }
    __syncthreads();

    const float* maas[5] = { maak, maaw, maav, maar, maag };
    for (int j = 0; j < 8; ++j) {
        int d = tid + 256 * j;
        float xl = xs[d], sx = sxs[d];
        float x5_1 = 0.f;
        for (int f = 0; f < 5; ++f) {
            float acc = 0.f;
            const float* w2f = w2 + ((size_t)f * LORAR) * DD + d;
            const float* t5f = t5s + f * LORAR;
            for (int r = 0; r < LORAR; ++r) acc += t5f[r] * w2f[(size_t)r * DD];
            float x5 = xl + sx * (maas[f][d] + acc);
            if (f == 1) x5_1 = x5;
            else        X5[((size_t)f * NTOK + n) * DD + d] = (bf16)x5;
        }
        xxs[d] = x5_1;   // xxs no longer needed; reuse for decay input
    }
    __syncthreads();

    if (tid < DLORA) {
        float acc = 0.f;
        for (int c = 0; c < DD; ++c) acc += xxs[c] * wd1[(size_t)c * DLORA + tid];
        decs[tid] = tanhf(acc);
    }
    __syncthreads();

    for (int j = 0; j < 8; ++j) {
        int d = tid + 256 * j;
        float acc = tdecay[d];
        for (int r = 0; r < DLORA; ++r) acc += decs[r] * wd2[(size_t)r * DD + d];
        wout[(size_t)n * DD + d] = expf(-expf(acc));
    }
}

// ---------------------------------------------------------------------------
// Tensor Data Mover: DMA one 128-row x 32-col bf16 tile (K-contiguous rows)
// into LDS with 16B row padding (LDS row stride = 40 bf16 = 80 B).
// D# per CDNA5 ISA:
//   group0: count=1 | lds_addr | global_addr[56:0] | type=2
//   group1: data_size=1(2B), pad_enable, pad_interval=3(16 DW), pad_amount=3(4 DW),
//           tensor_dim=(K,128), tile_dim=(32,128), dim0_stride=K
// ---------------------------------------------------------------------------
__device__ inline unsigned lds_off(const void* p) {
    return (unsigned)(uintptr_t)p;   // LDS aperture: low 32 bits of the pointer are the LDS address
}

__device__ inline void tdm_load_tile(const bf16* gsrc, unsigned ldsAddr, int K) {
    unsigned long long ga = (unsigned long long)(uintptr_t)gsrc;
    v4u g0;
    g0[0] = 1u;                                       // count = 1 valid descriptor
    g0[1] = ldsAddr;                                  // LDS byte address
    g0[2] = (unsigned)(ga & 0xFFFFFFFFu);             // global addr lo
    g0[3] = (unsigned)((ga >> 32) & 0x01FFFFFFu) | (2u << 30);   // hi + type=2
    v8i_t g1;
    g1[0] = (int)((1u << 16) |                        // data_size = 2 bytes
                  (1u << 20) |                        // pad_enable
                  (3u << 22) |                        // pad_interval: 16 DWORDs (one row)
                  (3u << 25));                        // pad_amount: 4 DWORDs (16 B)
    g1[1] = (int)(((unsigned)K & 0xFFFFu) << 16);     // tensor_dim0[15:0]
    g1[2] = (int)((((unsigned)K >> 16) & 0xFFFFu) |   // tensor_dim0[31:16]
                  (128u << 16));                      // tensor_dim1[15:0] = 128 rows
    g1[3] = (int)(32u << 16);                         // tensor_dim1[31:16]=0, tile_dim0=32
    g1[4] = (int)128u;                                // tile_dim1 = 128, tile_dim2 = 0
    g1[5] = (int)K;                                   // tensor_dim0_stride lo32
    g1[6] = 0;                                        // stride hi, dim1_stride lo
    g1[7] = 0;                                        // dim1_stride hi
    v4i_t z4 = { 0, 0, 0, 0 };
#if __has_include(<hip/amd_detail/amd_gfx1250_TDM.h>)
    v8i_t z8 = { 0, 0, 0, 0, 0, 0, 0, 0 };
    __builtin_amdgcn_tensor_load_to_lds(g0, g1, z4, z4, z8, 0);
#else
    __builtin_amdgcn_tensor_load_to_lds(g0, g1, z4, z4, 0);
#endif
}

// ---------------------------------------------------------------------------
// bf16 WMMA GEMM:  C[M,N] = A[M,K] * B[N,K]^T  (both operands K-contiguous).
// Block tile 128x128, 8 wave32 waves, wave tile 32x64 (2x4 wmma accs), K-step 32.
// Tiles staged by the Tensor Data Mover (wave 0 issues, TENSORcnt tracked),
// double-buffered in LDS -> one workgroup barrier per K-step.
// Fragment = two aligned ds_load_b128 matching the 16-bit A-matrix 16x32
// lane/K layout (CDNA5 ISA §7.12.2).
// ---------------------------------------------------------------------------
#define LDSW 40

__device__ inline v16bf load_frag(const bf16* rowp, int hi) {
    const U4* p = (const U4*)rowp;
    FragA f;
    f.u[0] = p[hi];       // K = hi*8 .. hi*8+7
    f.u[1] = p[2 + hi];   // K = 16 + hi*8 .. 16 + hi*8 + 7
    return f.v;
}

__global__ __launch_bounds__(256)
void gemm_bf16_nt(const bf16* __restrict__ A, const bf16* __restrict__ Bw,
                  float* __restrict__ C, int M, int N, int K)
{
    __shared__ bf16 As[2][128 * LDSW];
    __shared__ bf16 Bs[2][128 * LDSW];

    const int tid  = threadIdx.x;
    const int m0   = blockIdx.y * 128;
    const int n0   = blockIdx.x * 128;
    const int lane = tid & 31;
    const int w    = tid >> 5;
    const int wm   = w >> 1;          // 0..3 -> M offset wm*32
    const int wn   = w & 1;           // 0..1 -> N offset wn*64
    const int l15  = lane & 15;
    const int hi   = lane >> 4;

    v8f acc[2][4];
    for (int a = 0; a < 2; ++a)
        for (int bq = 0; bq < 4; ++bq)
            for (int e = 0; e < 8; ++e) acc[a][bq][e] = 0.f;

    const bf16* Abase = A  + (size_t)m0 * K;
    const bf16* Bbase = Bw + (size_t)n0 * K;

    if (w == 0) {
        tdm_load_tile(Abase, lds_off(&As[0][0]), K);
        tdm_load_tile(Bbase, lds_off(&Bs[0][0]), K);
        __builtin_amdgcn_s_wait_tensorcnt(0);
    }
    __syncthreads();

    int cur = 0;
    for (int k0 = 0; k0 < K; k0 += 32) {
        const bool more = (k0 + 32) < K;
        if (w == 0 && more) {   // kick next tile's DMA before computing
            tdm_load_tile(Abase + k0 + 32, lds_off(&As[cur ^ 1][0]), K);
            tdm_load_tile(Bbase + k0 + 32, lds_off(&Bs[cur ^ 1][0]), K);
        }

        v16bf aF[2], bF[4];
        for (int mt = 0; mt < 2; ++mt)
            aF[mt] = load_frag(&As[cur][(wm * 32 + mt * 16 + l15) * LDSW], hi);
        for (int nt = 0; nt < 4; ++nt)
            bF[nt] = load_frag(&Bs[cur][(wn * 64 + nt * 16 + l15) * LDSW], hi);

        for (int mt = 0; mt < 2; ++mt)
            for (int nt = 0; nt < 4; ++nt)
                acc[mt][nt] = __builtin_amdgcn_wmma_f32_16x16x32_bf16(
                    false, aF[mt], false, bF[nt], (short)0, acc[mt][nt], false, false);

        if (w == 0 && more) __builtin_amdgcn_s_wait_tensorcnt(0);
        __syncthreads();
        cur ^= 1;
    }

    // C/D layout: acc element r of lane l -> row (hi*8 + r), col l15
    for (int mt = 0; mt < 2; ++mt)
        for (int nt = 0; nt < 4; ++nt) {
            int mb = m0 + wm * 32 + mt * 16 + hi * 8;
            int nc = n0 + wn * 64 + nt * 16 + l15;
            for (int r = 0; r < 8; ++r)
                C[(size_t)(mb + r) * N + nc] = acc[mt][nt][r];
        }
}

// ---------------------------------------------------------------------------
// WKV recurrent scan: one block per (b,h).  256 threads = 4 row-groups x 64 cols.
// Each thread holds 16 state rows of its column in registers.
// ---------------------------------------------------------------------------
__global__ __launch_bounds__(256)
void wkv_kernel(const float* __restrict__ Kb, const float* __restrict__ Vb,
                const float* __restrict__ Rb, const float* __restrict__ Wd,
                const float* __restrict__ state, const float* __restrict__ faaaa,
                const int* __restrict__ ip, float* __restrict__ rwkv)
{
    __shared__ float kk[64], vv[64], rr[64], ww[64];
    __shared__ float part[4][64];

    const int bh = blockIdx.x;
    const int b  = bh / HH;
    const int h  = bh % HH;
    const int tid = threadIdx.x;
    const int j  = tid & 63;      // column
    const int ig = tid >> 6;      // row group 0..3
    const int i1 = STR * ip[0] + 1;

    float s[16], fa[16];
    for (int ii = 0; ii < 16; ++ii) {
        int i   = ig * 16 + ii;
        int idx = h * 4096 + i * 64 + j;        // flatten of (H,HS,HS)
        int row = idx >> 11;                    // back to state row (64 x D layout)
        int d   = idx & 2047;
        s[ii]  = state[((size_t)b * STR + i1 + 1 + row) * DD + d];
        fa[ii] = faaaa[h * 64 + i];
    }

    for (int t = 0; t < TT; ++t) {
        size_t base = ((size_t)(b * TT + t)) * DD + h * 64;
        {
            int arr = tid >> 6, ln = tid & 63;
            if      (arr == 0) kk[ln] = Kb[base + ln];
            else if (arr == 1) vv[ln] = Vb[base + ln];
            else if (arr == 2) rr[ln] = Rb[base + ln];
            else               ww[ln] = Wd[base + ln];
        }
        __syncthreads();

        float vj = vv[j];
        float partial = 0.f;
        for (int ii = 0; ii < 16; ++ii) {
            int i = ig * 16 + ii;
            float kv = kk[i] * vj;
            partial += rr[i] * (fa[ii] * kv + s[ii]);
            s[ii] = kv + ww[i] * s[ii];
        }
        part[ig][j] = partial;
        __syncthreads();
        if (ig == 0)
            rwkv[base + j] = part[0][j] + part[1][j] + part[2][j] + part[3][j];
        __syncthreads();
    }
}

// ---------------------------------------------------------------------------
// Per-head GroupNorm * silu(gate) -> bf16 operand for the output projection.
// ---------------------------------------------------------------------------
__global__ __launch_bounds__(256)
void gn_gate_kernel(const float* __restrict__ rwkv, const float* __restrict__ G,
                    const float* __restrict__ lnx_w, const float* __restrict__ lnx_b,
                    bf16* __restrict__ ng)
{
    __shared__ float rv[DD];
    __shared__ float mh[HH], vh[HH];
    const int n = blockIdx.x, tid = threadIdx.x;

    for (int j = 0; j < 8; ++j) { int d = tid + 256 * j; rv[d] = rwkv[(size_t)n * DD + d]; }
    __syncthreads();
    if (tid < HH) {
        float s = 0.f, s2 = 0.f;
        for (int e = 0; e < HSZ; ++e) { float v = rv[tid * HSZ + e]; s += v; s2 += v * v; }
        float m = s / HSZ;
        mh[tid] = m;
        vh[tid] = s2 / HSZ - m * m;
    }
    __syncthreads();
    for (int j = 0; j < 8; ++j) {
        int d = tid + 256 * j;
        int h = d >> 6;
        float val = (rv[d] - mh[h]) * rsqrtf(vh[h] + EPSV) * lnx_w[d] + lnx_b[d];
        float g   = G[(size_t)n * DD + d];
        float sil = g / (1.f + expf(-g));
        ng[(size_t)n * DD + d] = (bf16)(val * sil);
    }
}

// ---------------------------------------------------------------------------
extern "C" void kernel_launch(void* const* d_in, const int* in_sizes, int n_in,
                              void* d_out, int out_size, void* d_ws, size_t ws_size,
                              hipStream_t stream)
{
    const float* x        = (const float*)d_in[0];
    const float* state    = (const float*)d_in[1];
    const float* ln1_w    = (const float*)d_in[2];
    const float* ln1_b    = (const float*)d_in[3];
    const float* tmx      = (const float*)d_in[4];
    const float* tm_w1    = (const float*)d_in[5];
    const float* tm_w2    = (const float*)d_in[6];
    const float* maak     = (const float*)d_in[7];
    const float* maaw     = (const float*)d_in[8];
    const float* maav     = (const float*)d_in[9];
    const float* maar     = (const float*)d_in[10];
    const float* maag     = (const float*)d_in[11];
    const float* tdecay   = (const float*)d_in[12];
    const float* td_w1    = (const float*)d_in[13];
    const float* td_w2    = (const float*)d_in[14];
    const float* faaaa    = (const float*)d_in[15];
    const float* Wrec     = (const float*)d_in[16];
    const float* Wkey     = (const float*)d_in[17];
    const float* Wval     = (const float*)d_in[18];
    const float* Wout     = (const float*)d_in[19];
    const float* Wgate    = (const float*)d_in[20];
    const float* lnx_w    = (const float*)d_in[21];
    const float* lnx_b    = (const float*)d_in[22];
    const int*   ip       = (const int*)d_in[23];

    const size_t D2 = (size_t)DD * DD;       // 4,194,304
    const size_t ND = (size_t)NTOK * DD;     // 1,048,576

    char* ws = (char*)d_ws;
    bf16*  Wk_b  = (bf16*)ws;                //  8 MB each
    bf16*  Wv_b  = Wk_b + D2;
    bf16*  Wr_b  = Wv_b + D2;
    bf16*  Wg_b  = Wr_b + D2;
    bf16*  Wo_b  = Wg_b + D2;
    bf16*  X5    = Wo_b + D2;                // 5 * ND bf16 (10 MB)
    float* wdec  = (float*)(X5 + 5 * ND);    // 4 MB
    float* Kb    = wdec + ND;
    float* Vb    = Kb + ND;
    float* Rb    = Vb + ND;
    float* Gb    = Rb + ND;
    float* rwkvb = Gb + ND;
    bf16*  ng    = (bf16*)(rwkvb + ND);      // 2 MB

    // 1) weights -> bf16
    const int cvtBlocks = (int)(D2 / 1024);
    cvt_bf16_kernel<<<cvtBlocks, 256, 0, stream>>>(Wkey,  Wk_b, (int)D2);
    cvt_bf16_kernel<<<cvtBlocks, 256, 0, stream>>>(Wval,  Wv_b, (int)D2);
    cvt_bf16_kernel<<<cvtBlocks, 256, 0, stream>>>(Wrec,  Wr_b, (int)D2);
    cvt_bf16_kernel<<<cvtBlocks, 256, 0, stream>>>(Wgate, Wg_b, (int)D2);
    cvt_bf16_kernel<<<cvtBlocks, 256, 0, stream>>>(Wout,  Wo_b, (int)D2);

    // 2) LN + shift + LoRAs
    prep_kernel<<<NTOK, 256, 0, stream>>>(x, state, ln1_w, ln1_b, tmx, tm_w1, tm_w2,
                                          maak, maaw, maav, maar, maag,
                                          tdecay, td_w1, td_w2, ip, X5, wdec);

    // 3) the four projection GEMMs (WMMA bf16, TDM-staged tiles)
    dim3 gg(DD / 128, NTOK / 128);           // (16, 4)
    gemm_bf16_nt<<<gg, 256, 0, stream>>>(X5 + 0 * ND, Wk_b, Kb, NTOK, DD, DD);
    gemm_bf16_nt<<<gg, 256, 0, stream>>>(X5 + 2 * ND, Wv_b, Vb, NTOK, DD, DD);
    gemm_bf16_nt<<<gg, 256, 0, stream>>>(X5 + 3 * ND, Wr_b, Rb, NTOK, DD, DD);
    gemm_bf16_nt<<<gg, 256, 0, stream>>>(X5 + 4 * ND, Wg_b, Gb, NTOK, DD, DD);

    // 4) sequential WKV scan
    wkv_kernel<<<BB * HH, 256, 0, stream>>>(Kb, Vb, Rb, wdec, state, faaaa, ip, rwkvb);

    // 5) GroupNorm * silu(gate)
    gn_gate_kernel<<<NTOK, 256, 0, stream>>>(rwkvb, Gb, lnx_w, lnx_b, ng);

    // 6) output projection -> fp32 d_out
    gemm_bf16_nt<<<gg, 256, 0, stream>>>(ng, Wo_b, (float*)d_out, NTOK, DD, DD);
}
